// MoELinear_19378892440176
// MI455X (gfx1250) — matrix operand order repaired and checked by reading, not compiled, not run on hardware
//
#include <hip/hip_runtime.h>

// MoE ragged grouped GEMM for gfx1250 (MI455X).
// fp32 inputs/weights staged to LDS as bf16 (RNE), fp32 accumulation via
// v_wmma_f32_16x16x32_bf16, fp32 bias + output.
// Block tile 128x128 (8 wave32s in 4x2 grid, 32x64 per wave = 2x4 WMMA frags).
// Software pipeline: next K-tile is prefetched into VGPRs during the current
// tile's WMMA phase; single LDS buffer (36.9 KB).

namespace {
constexpr int kE   = 8;      // experts
constexpr int kTok = 8192;   // total tokens
constexpr int kI   = 2048;   // in features (contraction)
constexpr int kO   = 4096;   // out features
constexpr int BM   = 128;    // block tile M
constexpr int BN   = 128;    // block tile N
constexpr int KT   = 64;     // block tile K (two WMMA K=32 steps)
constexpr int LPAD = 8;      // LDS padding (bf16 elements)
constexpr int LSTR = KT + LPAD;              // 72 elems -> 144 B row stride
constexpr int MAX_TILES = kTok / BM + kE;    // 72 worst-case row tiles
}

typedef __attribute__((ext_vector_type(16))) __bf16 v16bf;
typedef __attribute__((ext_vector_type(2)))  __bf16 v2bf;
typedef __attribute__((ext_vector_type(2)))  float  v2f;
typedef __attribute__((ext_vector_type(8)))  float  v8f;
typedef __attribute__((ext_vector_type(8)))  unsigned int v8u;

__device__ __forceinline__ unsigned int pack2_bf16(float a, float b) {
  // Packed fp32 -> bf16 via vector fptrunc; lowers to the native packed
  // convert when the target has one (gfx1250 has full BF16 VALU support),
  // otherwise clang inline-expands an RNE sequence. a -> low half.
  const v2f src = {a, b};
  const v2bf dst = __builtin_convertvector(src, v2bf);
  return __builtin_bit_cast(unsigned int, dst);
}

__global__ __launch_bounds__(256) void moe_grouped_gemm_bf16w(
    const float* __restrict__ A,     // [T, I]
    const int*   __restrict__ gs,    // [E]
    const float* __restrict__ W,     // [E, I, O]
    const float* __restrict__ bias,  // [E, O]
    float* __restrict__ out)         // [T, O]
{
  __shared__ __align__(16) unsigned short Asm[BM * LSTR];
  __shared__ __align__(16) unsigned short Bsm[BN * LSTR];

  // ---- map blockIdx.x -> (expert, rowStart, rowsIn); fully block-uniform ----
  int idx = blockIdx.x;
  int expert = -1, rowStart = 0, rowsIn = 0, off = 0;
  #pragma unroll
  for (int e = 0; e < kE; ++e) {
    const int s  = gs[e];
    const int nt = (s + BM - 1) / BM;
    if (expert < 0) {
      if (idx < nt) {
        expert   = e;
        rowStart = off + idx * BM;
        const int rem = s - idx * BM;
        rowsIn   = rem < BM ? rem : BM;
      } else {
        idx -= nt;
      }
    }
    off += s;
  }
  if (expert < 0) return;  // uniform early exit (EXEC all-ones in active blocks)

  const int n0 = blockIdx.y * BN;
  const float* __restrict__ Wv = W + (size_t)expert * kI * kO;

  const int tid  = threadIdx.x;
  const int lane = tid & 31;
  const int wave = tid >> 5;
  const int wm   = wave & 3;   // wave row (0..3) -> 32 rows each
  const int wn   = wave >> 2;  // wave col (0..1) -> 64 cols each
  const int half = lane >> 4;  // lane group (0..1)
  const int l16  = lane & 15;

  v8f acc[2][4];
  #pragma unroll
  for (int i = 0; i < 2; ++i)
    #pragma unroll
    for (int j = 0; j < 4; ++j)
      acc[i][j] = (v8f){0.f, 0.f, 0.f, 0.f, 0.f, 0.f, 0.f, 0.f};

  // ---- register staging buffers (software pipeline, depth 2) ----
  float4 aR[8];           // A: 8 float4 per thread (128x64 tile)
  float4 bR0[4], bR1[4];  // B: 4 (k,k+1) float4-row pairs per thread

  // Per-thread staging coordinates (loop-invariant)
  // A: float4 f = tid + i*256 -> row r = f>>4, quad kq = f&15
  // B: pair   p = tid + i*256 -> k-pair kp = p>>5, col quad q = p&31
  auto load_tiles = [&](int k0) {
    #pragma unroll
    for (int i = 0; i < 8; ++i) {
      const int f  = tid + i * 256;
      const int r  = f >> 4;
      const int kq = f & 15;
      int g = rowStart + r;
      if (g > kTok - 1) g = kTok - 1;    // clamp (never stored)
      aR[i] = *(const float4*)(A + (size_t)g * kI + k0 + kq * 4);
    }
    #pragma unroll
    for (int i = 0; i < 4; ++i) {
      const int p  = tid + i * 256;
      const int kp = p >> 5;
      const int q  = p & 31;
      const int k  = k0 + 2 * kp;
      bR0[i] = *(const float4*)(Wv + (size_t)k * kO + n0 + 4 * q);
      bR1[i] = *(const float4*)(Wv + (size_t)(k + 1) * kO + n0 + 4 * q);
    }
  };

  auto store_tiles = [&]() {
    #pragma unroll
    for (int i = 0; i < 8; ++i) {
      const int f  = tid + i * 256;
      const int r  = f >> 4;
      const int kq = f & 15;
      *(uint2*)((char*)Asm + r * (LSTR * 2) + kq * 8) =
          make_uint2(pack2_bf16(aR[i].x, aR[i].y),
                     pack2_bf16(aR[i].z, aR[i].w));
    }
    #pragma unroll
    for (int i = 0; i < 4; ++i) {
      const int p  = tid + i * 256;
      const int kp = p >> 5;
      const int q  = p & 31;
      char* bp = (char*)Bsm + (4 * q) * (LSTR * 2) + (2 * kp) * 2;
      *(unsigned int*)(bp)                = pack2_bf16(bR0[i].x, bR1[i].x);
      *(unsigned int*)(bp + LSTR * 2)     = pack2_bf16(bR0[i].y, bR1[i].y);
      *(unsigned int*)(bp + 2 * LSTR * 2) = pack2_bf16(bR0[i].z, bR1[i].z);
      *(unsigned int*)(bp + 3 * LSTR * 2) = pack2_bf16(bR0[i].w, bR1[i].w);
    }
  };

  load_tiles(0);  // prologue

  for (int k0 = 0; k0 < kI; k0 += KT) {
    // ---- convert + store the prefetched tile to LDS ----
    store_tiles();
    __syncthreads();

    // ---- issue global prefetch of the next K-tile (hidden under WMMAs) ----
    if (k0 + KT < kI) load_tiles(k0 + KT);  // uniform branch

    // ---- 2 x (K=32) WMMA steps over the staged tiles ----
    #pragma unroll
    for (int kk = 0; kk < KT; kk += 32) {
      v16bf afrag[2], bfrag[4];
      #pragma unroll
      for (int mi = 0; mi < 2; ++mi) {
        // A 16x32 bf16 layout: lane = M row, elems 0-7 at K=half*8,
        // elems 8-15 at K=16+half*8 (ISA 7.12.2)
        const int  row = wm * 32 + mi * 16 + l16;
        const char* pa = (const char*)Asm + row * (LSTR * 2);
        const uint4 c0 = *(const uint4*)(pa + (kk + half * 8) * 2);
        const uint4 c1 = *(const uint4*)(pa + (kk + 16 + half * 8) * 2);
        const v8u u = {c0.x, c0.y, c0.z, c0.w, c1.x, c1.y, c1.z, c1.w};
        afrag[mi] = __builtin_bit_cast(v16bf, u);
      }
      #pragma unroll
      for (int ni = 0; ni < 4; ++ni) {
        // B 32x16 bf16 layout: lane = N col, elems 0-7 at K=half*8,
        // elems 8-15 at K=16+half*8
        const int  col = wn * 64 + ni * 16 + l16;
        const char* pb = (const char*)Bsm + col * (LSTR * 2) + (kk + half * 8) * 2;
        const uint4 c0 = *(const uint4*)(pb);
        const uint4 c1 = *(const uint4*)(pb + 32);
        const v8u u = {c0.x, c0.y, c0.z, c0.w, c1.x, c1.y, c1.z, c1.w};
        bfrag[ni] = __builtin_bit_cast(v16bf, u);
      }
      #pragma unroll
      for (int mi = 0; mi < 2; ++mi)
        #pragma unroll
        for (int ni = 0; ni < 4; ++ni)
          acc[mi][ni] = __builtin_amdgcn_wmma_f32_16x16x32_bf16(
              false, afrag[mi], false, bfrag[ni],
              (short)0, acc[mi][ni], false, false);
    }
    __syncthreads();
  }

  // ---- epilogue: bias add + store ----
  // C/D layout: VGPR r -> M = r + 8*half; N = lane%16 (ISA 7.12.2)
  float bv[4];
  #pragma unroll
  for (int ni = 0; ni < 4; ++ni)
    bv[ni] = bias[expert * kO + n0 + wn * 64 + ni * 16 + l16];

  if (rowsIn == BM) {
    // Full tile: branch-free store path (uniform condition, no exec churn)
    #pragma unroll
    for (int mi = 0; mi < 2; ++mi) {
      const int rbase = wm * 32 + mi * 16 + half * 8;
      #pragma unroll
      for (int ni = 0; ni < 4; ++ni) {
        const int coln = n0 + wn * 64 + ni * 16 + l16;
        float* op = out + (size_t)(rowStart + rbase) * kO + coln;
        #pragma unroll
        for (int r = 0; r < 8; ++r)
          op[(size_t)r * kO] = acc[mi][ni][r] + bv[ni];
      }
    }
  } else {
    // Ragged tail tile: per-row masked stores
    #pragma unroll
    for (int mi = 0; mi < 2; ++mi) {
      const int rbase = wm * 32 + mi * 16 + half * 8;
      #pragma unroll
      for (int ni = 0; ni < 4; ++ni) {
        const int coln = n0 + wn * 64 + ni * 16 + l16;
        #pragma unroll
        for (int r = 0; r < 8; ++r) {
          const int lr = rbase + r;
          if (lr < rowsIn)
            out[(size_t)(rowStart + lr) * kO + coln] = acc[mi][ni][r] + bv[ni];
        }
      }
    }
  }
}

extern "C" void kernel_launch(void* const* d_in, const int* in_sizes, int n_in,
                              void* d_out, int out_size, void* d_ws, size_t ws_size,
                              hipStream_t stream) {
  (void)in_sizes; (void)n_in; (void)d_ws; (void)ws_size; (void)out_size;
  const float* inputs      = (const float*)d_in[0];
  const int*   group_sizes = (const int*)d_in[1];
  const float* kernelw     = (const float*)d_in[2];
  const float* bias        = (const float*)d_in[3];
  float*       out         = (float*)d_out;

  dim3 grid(MAX_TILES, kO / BN);  // 72 x 32; inactive row-tiles exit uniformly
  dim3 block(256);                // 8 wave32s
  moe_grouped_gemm_bf16w<<<grid, block, 0, stream>>>(inputs, group_sizes,
                                                     kernelw, bias, out);
}
